// RASM_64793876627727
// MI455X (gfx1250) — matrix operand order repaired and checked by reading, not compiled, not run on hardware
//
#include <hip/hip_runtime.h>
#include <hip/hip_bf16.h>
#include <stdint.h>

// ---------------------------------------------------------------------------
// Types for CDNA5 WMMA (wave32) and TDM descriptors
// ---------------------------------------------------------------------------
typedef __bf16 bf16_t;
typedef __attribute__((ext_vector_type(16))) __bf16 v16bf;
typedef __attribute__((ext_vector_type(8)))  __bf16 v8bf;
typedef __attribute__((ext_vector_type(8)))  float  v8f;
typedef __attribute__((ext_vector_type(4)))  float  v4f;
typedef unsigned int u32x4 __attribute__((ext_vector_type(4)));
typedef int          i32x8 __attribute__((ext_vector_type(8)));
typedef int          i32x4 __attribute__((ext_vector_type(4)));

#define NBATCH   8
#define NSEQ     256
#define NSEQT    260          // N_PERSIST + SEQ
#define NPERSIST 4
#define DMODEL   512
#define DSTATE   128
#define NVOCAB   32000
#define DHID     64
#define DCTRL    640          // DSTATE + DMODEL
#define LN_EPS   1e-5f

// ---------------------------------------------------------------------------
// Device helpers
// ---------------------------------------------------------------------------
__device__ __forceinline__ float sigmoidf_(float x) { return 1.0f / (1.0f + __expf(-x)); }
__device__ __forceinline__ float siluf_(float x)    { return x * sigmoidf_(x); }

__device__ __forceinline__ float block_sum256(float v, float* red) {
    const int tid = threadIdx.x;
    red[tid] = v;
    __syncthreads();
    #pragma unroll
    for (int o = 128; o > 0; o >>= 1) {
        if (tid < o) red[tid] += red[tid + o];
        __syncthreads();
    }
    float r = red[0];
    __syncthreads();
    return r;
}

// Load a 16x32 bf16 A-matrix fragment (row-major source, stride in elements)
// per the CDNA5 16-bit A layout:
//   lanes 0-15 : m = lane,    K = {0..7, 16..23}
//   lanes16-31 : m = lane-16, K = {8..15, 24..31}
__device__ __forceinline__ v16bf load_afrag(const bf16_t* __restrict__ base,
                                            int row, int stride, int kbase, int lane) {
    const bf16_t* rp = base + (size_t)row * stride + kbase;
    const int off = (lane < 16) ? 0 : 8;
    v8bf lo = *(const v8bf*)(rp + off);        // K = off .. off+7
    v8bf hi = *(const v8bf*)(rp + off + 16);   // K = off+16 .. off+23
    v16bf a;
    #pragma unroll
    for (int i = 0; i < 8; ++i) { a[i] = lo[i]; a[8 + i] = hi[i]; }
    return a;
}

// TDM: 1D contiguous copy of `nelem` bf16 (2B) elements global -> LDS.
// D# per CDNA5 ISA (group0/group1; groups 2/3 unused).
// This toolchain declares the 6-arg builtin (clang-23 / therock-10.0 form).
__device__ __forceinline__ void tdm_load_1d(unsigned lds_off, const void* gsrc, unsigned nelem) {
    const unsigned long long ga = (unsigned long long)(uintptr_t)gsrc;
    u32x4 g0;
    g0[0] = 1u;                                         // count=1, user descriptor
    g0[1] = lds_off;                                    // lds_addr (bytes)
    g0[2] = (unsigned)(ga & 0xFFFFFFFFu);               // global_addr[31:0]
    g0[3] = (unsigned)((ga >> 32) & 0x01FFFFFFu)        // global_addr[56:32]
          | (2u << 30);                                 // type = 2 ("image")
    i32x8 g1;
    g1[0] = (int)(1u << 16);                            // data_size=1 (2 bytes), no multicast
    g1[1] = (int)((nelem & 0xFFFFu) << 16);             // tensor_dim0[15:0]
    g1[2] = (int)(((nelem >> 16) & 0xFFFFu)             // tensor_dim0[31:16]
          | (1u << 16));                                // tensor_dim1 = 1
    g1[3] = (int)((nelem & 0xFFFFu) << 16);             // tile_dim0 = nelem (<=65535)
    g1[4] = 1;                                          // tile_dim1 = 1, tile_dim2 = 0
    g1[5] = (int)nelem;                                 // tensor_dim0_stride[31:0]
    g1[6] = 0;                                          // stride hi / dim1_stride lo
    g1[7] = 0;
    i32x4 gz4 = { 0, 0, 0, 0 };
    i32x8 gz8 = { 0, 0, 0, 0, 0, 0, 0, 0 };
    __builtin_amdgcn_tensor_load_to_lds(g0, g1, gz4, gz4, gz8, 0);
}

// ---------------------------------------------------------------------------
// One-time kernels
// ---------------------------------------------------------------------------
__global__ void init_state_kernel(float* __restrict__ h, float* __restrict__ Aflat,
                                  float* __restrict__ Bflat,
                                  bf16_t* __restrict__ hidAbf, bf16_t* __restrict__ hidBbf) {
    long idx = (long)blockIdx.x * 256 + threadIdx.x;
    if (idx < NBATCH * DSTATE) h[idx] = 0.0f;
    if (idx < 16 * DHID) { hidAbf[idx] = (bf16_t)0.0f; hidBbf[idx] = (bf16_t)0.0f; }
    if (idx < (long)NBATCH * DSTATE * DSTATE) {
        int n = (int)(idx % (DSTATE * DSTATE));
        Aflat[idx] = ((n >> 7) == (n & 127)) ? 1.0f : 0.0f;   // identity
    }
    if (idx < (long)NBATCH * DSTATE * DMODEL) Bflat[idx] = 0.0f;
}

__global__ void embed_kernel(const int* __restrict__ ids, const float* __restrict__ emb,
                             const float* __restrict__ persist, float* __restrict__ xseq) {
    long idx = (long)blockIdx.x * 256 + threadIdx.x;
    if (idx >= (long)NSEQT * NBATCH * DMODEL) return;
    int d = (int)(idx % DMODEL);
    long r = idx / DMODEL;
    int b = (int)(r % NBATCH);
    int t = (int)(r / NBATCH);
    float v;
    if (t < NPERSIST) v = persist[t * DMODEL + d];
    else              v = emb[(size_t)ids[b * NSEQ + (t - NPERSIST)] * DMODEL + d];
    xseq[idx] = v;
}

// Pre-pack W (K x N, row-major f32) into bf16 WMMA B-fragments:
// out[(tile*KC + kc)*512 + lane*16 + e] = W[k][n]
//   n = tile*16 + lane%16 ; k = kc*32 + 8*(lane/16) + (e&7) + 16*(e>>3)   (dense 16-bit B layout)
__global__ void pack_bfrag_kernel(const float* __restrict__ W, bf16_t* __restrict__ out,
                                  int N, int K) {
    long o = (long)blockIdx.x * 256 + threadIdx.x;
    long total = (long)N * K;
    if (o >= total) return;
    int  KC = K >> 5;
    long per_tile = (long)KC * 512;
    long tile = o / per_tile;
    int  r  = (int)(o % per_tile);
    int  kc = r >> 9;
    int  q  = r & 511;
    int  lane = q >> 4;
    int  e    = q & 15;
    int  n = (int)tile * 16 + (lane & 15);
    int  k = kc * 32 + ((lane >> 4) << 3) + (e & 7) + ((e >> 3) << 4);
    out[o] = (bf16_t)W[(size_t)k * N + n];
}

// ---------------------------------------------------------------------------
// Per-step kernel 1: ctrl LayerNorm + all six layer-1 MLPs + gates/rate
// ---------------------------------------------------------------------------
__global__ void ctrl_mlp_kernel(
    const float* __restrict__ xseq, const float* __restrict__ h,
    const float* __restrict__ lncg, const float* __restrict__ lncb,
    const float* __restrict__ rW1, const float* __restrict__ rb1,
    const float* __restrict__ rW2, const float* __restrict__ rb2,
    const float* __restrict__ aW1, const float* __restrict__ ab1,
    const float* __restrict__ bW1, const float* __restrict__ bb1,
    const float* __restrict__ tW1, const float* __restrict__ tb1,
    const float* __restrict__ tW2, const float* __restrict__ tb2,
    const float* __restrict__ fW1, const float* __restrict__ fb1,
    const float* __restrict__ fW2, const float* __restrict__ fb2,
    const float* __restrict__ uW1, const float* __restrict__ ub1,
    const float* __restrict__ uW2, const float* __restrict__ ub2,
    bf16_t* __restrict__ hidAbf, bf16_t* __restrict__ hidBbf,
    float* __restrict__ rate, float* __restrict__ fg, float* __restrict__ ug,
    int t) {
    __shared__ float c[DCTRL];
    __shared__ float hid[6 * DHID];   // 0 router, 1 hypA, 2 hypB, 3 rate, 4 forget, 5 update
    __shared__ float red[256];

    const int b = blockIdx.x, tid = threadIdx.x;
    const float* xt = xseq + ((size_t)t * NBATCH + b) * DMODEL;

    for (int k = tid; k < DCTRL; k += 256)
        c[k] = (k < DSTATE) ? h[b * DSTATE + k] : xt[k - DSTATE];
    __syncthreads();

    float s = 0.0f;
    for (int k = tid; k < DCTRL; k += 256) s += c[k];
    const float mean = block_sum256(s, red) * (1.0f / DCTRL);
    float vs = 0.0f;
    for (int k = tid; k < DCTRL; k += 256) { float d = c[k] - mean; vs += d * d; }
    const float var = block_sum256(vs, red) * (1.0f / DCTRL);
    const float rsq = rsqrtf(var + LN_EPS);
    for (int k = tid; k < DCTRL; k += 256)
        c[k] = (c[k] - mean) * rsq * lncg[k] + lncb[k];
    __syncthreads();

    const float* W1s[6] = { rW1, aW1, bW1, tW1, fW1, uW1 };
    const float* b1s[6] = { rb1, ab1, bb1, tb1, fb1, ub1 };
    for (int o = tid; o < 6 * DHID; o += 256) {
        const int m = o >> 6, j = o & 63;
        const float* W = W1s[m];
        float acc = b1s[m][j];
        for (int k = 0; k < DCTRL; ++k) acc += c[k] * W[k * DHID + j];
        hid[o] = siluf_(acc);
    }
    __syncthreads();

    if (tid < DHID) {
        hidAbf[b * DHID + tid] = (bf16_t)hid[1 * DHID + tid];
        hidBbf[b * DHID + tid] = (bf16_t)hid[2 * DHID + tid];
    }
    if (tid < DSTATE) {
        float a = fb2[tid];
        for (int k = 0; k < DHID; ++k) a += hid[4 * DHID + k] * fW2[k * DSTATE + tid];
        fg[b * DSTATE + tid] = sigmoidf_(a);
    } else if (tid < 2 * DSTATE) {
        const int i = tid - DSTATE;
        float a = ub2[i];
        for (int k = 0; k < DHID; ++k) a += hid[5 * DHID + k] * uW2[k * DSTATE + i];
        ug[b * DSTATE + i] = sigmoidf_(a);
    }
    if (tid == 0) {
        float r1 = rb2[0], r2 = tb2[0];
        for (int k = 0; k < DHID; ++k) { r1 += hid[0 * DHID + k] * rW2[k]; r2 += hid[3 * DHID + k] * tW2[k]; }
        rate[b] = sigmoidf_(r1) * sigmoidf_(r2);   // sigmoid(rate)*adapt_prob
    }
}

// ---------------------------------------------------------------------------
// Per-step kernel 2: hypernet GEMM + fused state-matrix update.
// hid (16x64 bf16, 2KB) is staged into LDS once per block by the TDM, then
// every wave computes one 16-wide N tile with two v_wmma_f32_16x16x32_bf16.
// NFLAT is a compile-time constant so the 8 per-batch rows become immediate
// IOFFSETs -> clause of 8 loads / one wait / 8 FMAs / clause of 8 stores.
// ---------------------------------------------------------------------------
template <int NFLAT>
__global__ void hyp_wmma_kernel(const bf16_t* __restrict__ hidbf,
                                const bf16_t* __restrict__ pW2,
                                const float*  __restrict__ b2,
                                float*        __restrict__ target,
                                const float*  __restrict__ rate) {
    __shared__ bf16_t hloc[16 * DHID];
    const int lane = threadIdx.x & 31;
    const int wv   = threadIdx.x >> 5;

    if (wv == 0) {  // uniform per-wave branch: TDM issued by wave 0 only
        tdm_load_1d((unsigned)(uintptr_t)hloc, hidbf, 16 * DHID);
        __builtin_amdgcn_s_wait_tensorcnt(0);
    }
    __syncthreads();

    const int tile = blockIdx.x * 8 + wv;
    v16bf a0 = load_afrag(hloc, lane & 15, DHID, 0,  lane);
    v16bf a1 = load_afrag(hloc, lane & 15, DHID, 32, lane);
    const bf16_t* bp = pW2 + (size_t)tile * 1024 + lane * 16;
    v16bf b0 = *(const v16bf*)bp;
    v16bf b1 = *(const v16bf*)(bp + 512);

    v8f acc = {};
    acc = __builtin_amdgcn_wmma_f32_16x16x32_bf16(false, a0, false, b0, (short)0, acc, false, false);
    acc = __builtin_amdgcn_wmma_f32_16x16x32_bf16(false, a1, false, b1, (short)0, acc, false, false);

    if (lane < 16) {  // D layout: lanes 0-15 / VGPR v hold batches 0-7; 8-15 are pad
        const int n = tile * 16 + lane;
        const float bias = b2[n];
        v4f r0 = *(const v4f*)(rate);
        v4f r1 = *(const v4f*)(rate + 4);
        float rr[8] = { r0[0], r0[1], r0[2], r0[3], r1[0], r1[1], r1[2], r1[3] };
        float* tp = target + n;
        float oldv[8];
        #pragma unroll
        for (int v = 0; v < 8; ++v) oldv[v] = tp[(size_t)v * NFLAT];
        #pragma unroll
        for (int v = 0; v < 8; ++v)
            tp[(size_t)v * NFLAT] = oldv[v] + rr[v] * (acc[v] + bias);
    }
}

// ---------------------------------------------------------------------------
// Per-step kernel 3: h = LN(f*(A@h) + u*(B@x)) ; y = LN(h @ C^T) -> bf16
// ---------------------------------------------------------------------------
__global__ void state_out_kernel(const float* __restrict__ Aflat, const float* __restrict__ Bflat,
                                 const float* __restrict__ xseq, float* __restrict__ h,
                                 const float* __restrict__ fg, const float* __restrict__ ug,
                                 const float* __restrict__ lnsg, const float* __restrict__ lnsb,
                                 const float* __restrict__ Cmat,
                                 const float* __restrict__ lnog, const float* __restrict__ lnob,
                                 bf16_t* __restrict__ ysbf, int t) {
    __shared__ float hv[DSTATE], xv[DMODEL], sv[DSTATE], hn[DSTATE], yv[DMODEL], red[256];
    const int b = blockIdx.x, tid = threadIdx.x;
    const float* xt = xseq + ((size_t)t * NBATCH + b) * DMODEL;

    if (tid < DSTATE) hv[tid] = h[b * DSTATE + tid];
    xv[tid]       = xt[tid];
    xv[tid + 256] = xt[tid + 256];
    __syncthreads();

    if (tid < DSTATE) {
        const float* Ar = Aflat + (size_t)b * (DSTATE * DSTATE) + (size_t)tid * DSTATE;
        float ah = 0.0f;
        for (int j = 0; j < DSTATE; ++j) ah += Ar[j] * hv[j];
        const float* Br = Bflat + (size_t)b * (DSTATE * DMODEL) + (size_t)tid * DMODEL;
        float bx = 0.0f;
        for (int d = 0; d < DMODEL; ++d) bx += Br[d] * xv[d];
        sv[tid] = fg[b * DSTATE + tid] * ah + ug[b * DSTATE + tid] * bx;
    }
    __syncthreads();

    float s = (tid < DSTATE) ? sv[tid] : 0.0f;
    const float mean = block_sum256(s, red) * (1.0f / DSTATE);
    float dv = (tid < DSTATE) ? (sv[tid] - mean) : 0.0f;
    const float var = block_sum256(dv * dv, red) * (1.0f / DSTATE);
    const float rsq = rsqrtf(var + LN_EPS);
    if (tid < DSTATE) {
        const float hh = (sv[tid] - mean) * rsq * lnsg[tid] + lnsb[tid];
        hn[tid] = hh;
        h[b * DSTATE + tid] = hh;
    }
    __syncthreads();

    if (t >= NPERSIST) {
        for (int d = tid; d < DMODEL; d += 256) {
            const float* Cr = Cmat + (size_t)d * DSTATE;
            float y = 0.0f;
            for (int i = 0; i < DSTATE; ++i) y += hn[i] * Cr[i];
            yv[d] = y;
        }
        __syncthreads();
        const float m2 = block_sum256(yv[tid] + yv[tid + 256], red) * (1.0f / DMODEL);
        const float e0 = yv[tid] - m2, e1 = yv[tid + 256] - m2;
        const float v2 = block_sum256(e0 * e0 + e1 * e1, red) * (1.0f / DMODEL);
        const float rs2 = rsqrtf(v2 + LN_EPS);
        bf16_t* yrow = ysbf + ((size_t)b * NSEQ + (t - NPERSIST)) * DMODEL;
        for (int d = tid; d < DMODEL; d += 256)
            yrow[d] = (bf16_t)((yv[d] - m2) * rs2 * lnog[d] + lnob[d]);
    }
}

// ---------------------------------------------------------------------------
// Final logits GEMM: (2048 x 512) @ (512 x 32000) + bias.
// Block = one 16-row M strip: TDM stages the 16x512 bf16 A strip (16 KB) into
// LDS once; each of the 8 waves computes a 16x64 output strip (4 N tiles),
// K fully unrolled -> 64 v_wmma_f32_16x16x32_bf16 per wave, 4 WMMA per A-frag.
// ---------------------------------------------------------------------------
__global__ void logits_wmma_kernel(const bf16_t* __restrict__ ysbf,
                                   const bf16_t* __restrict__ pWout,
                                   const float*  __restrict__ bout,
                                   float*        __restrict__ out) {
    __shared__ bf16_t atile[16 * DMODEL];   // 16 KB
    const int lane = threadIdx.x & 31;
    const int wv   = threadIdx.x >> 5;
    const int mt   = blockIdx.x & 127;      // 128 M tiles
    const int gidx = blockIdx.x >> 7;       // 63 groups of 8 waves -> 504 >= 500

    if (wv == 0) {
        tdm_load_1d((unsigned)(uintptr_t)atile, ysbf + (size_t)mt * 16 * DMODEL, 16 * DMODEL);
        __builtin_amdgcn_s_wait_tensorcnt(0);
    }
    __syncthreads();

    const int ngrp = gidx * 8 + wv;         // 4-tile group index, 0..499 valid
    if (ngrp < (NVOCAB / 16) / 4) {
        v8f acc0 = {}, acc1 = {}, acc2 = {}, acc3 = {};
        const bf16_t* bbase = pWout + (size_t)(ngrp * 4) * (16 * 512) + lane * 16;
        #pragma unroll
        for (int kc = 0; kc < 16; ++kc) {
            v16bf a  = load_afrag(atile, lane & 15, DMODEL, kc * 32, lane);
            const bf16_t* bq = bbase + (size_t)kc * 512;
            v16bf b0 = *(const v16bf*)(bq);
            v16bf b1 = *(const v16bf*)(bq + 1 * (16 * 512));
            v16bf b2 = *(const v16bf*)(bq + 2 * (16 * 512));
            v16bf b3 = *(const v16bf*)(bq + 3 * (16 * 512));
            if (kc < 15) __builtin_prefetch((const void*)(bq + 512), 0, 1);
            acc0 = __builtin_amdgcn_wmma_f32_16x16x32_bf16(false, a, false, b0, (short)0, acc0, false, false);
            acc1 = __builtin_amdgcn_wmma_f32_16x16x32_bf16(false, a, false, b1, (short)0, acc1, false, false);
            acc2 = __builtin_amdgcn_wmma_f32_16x16x32_bf16(false, a, false, b2, (short)0, acc2, false, false);
            acc3 = __builtin_amdgcn_wmma_f32_16x16x32_bf16(false, a, false, b3, (short)0, acc3, false, false);
        }
        const int mrow = mt * 16 + ((lane >= 16) ? 8 : 0);
        v8f accs[4] = { acc0, acc1, acc2, acc3 };
        #pragma unroll
        for (int nn = 0; nn < 4; ++nn) {
            const int col = (ngrp * 4 + nn) * 16 + (lane & 15);
            const float bias = bout[col];
            #pragma unroll
            for (int v = 0; v < 8; ++v)
                out[(size_t)(mrow + v) * NVOCAB + col] = accs[nn][v] + bias;
        }
    }
}

__global__ void hfinal_kernel(const float* __restrict__ h, float* __restrict__ out) {
    const int i = blockIdx.x * 256 + threadIdx.x;
    if (i < NBATCH * DSTATE) out[i] = h[i];
}

// ---------------------------------------------------------------------------
// Host launcher
// ---------------------------------------------------------------------------
extern "C" void kernel_launch(void* const* d_in, const int* in_sizes, int n_in,
                              void* d_out, int out_size, void* d_ws, size_t ws_size,
                              hipStream_t stream) {
    (void)in_sizes; (void)n_in; (void)out_size; (void)ws_size;

    const int*   ids     = (const int*)d_in[0];
    const float* emb     = (const float*)d_in[1];
    const float* persist = (const float*)d_in[2];
    const float* Cmat    = (const float*)d_in[3];
    const float* rW1 = (const float*)d_in[4],  *rb1 = (const float*)d_in[5];
    const float* rW2 = (const float*)d_in[6],  *rb2 = (const float*)d_in[7];
    const float* aW1 = (const float*)d_in[8],  *ab1 = (const float*)d_in[9];
    const float* aW2 = (const float*)d_in[10], *ab2 = (const float*)d_in[11];
    const float* bW1 = (const float*)d_in[12], *bb1 = (const float*)d_in[13];
    const float* bW2 = (const float*)d_in[14], *bb2 = (const float*)d_in[15];
    const float* tW1 = (const float*)d_in[16], *tb1 = (const float*)d_in[17];
    const float* tW2 = (const float*)d_in[18], *tb2 = (const float*)d_in[19];
    const float* fW1 = (const float*)d_in[20], *fb1 = (const float*)d_in[21];
    const float* fW2 = (const float*)d_in[22], *fb2 = (const float*)d_in[23];
    const float* uW1 = (const float*)d_in[24], *ub1 = (const float*)d_in[25];
    const float* uW2 = (const float*)d_in[26], *ub2 = (const float*)d_in[27];
    const float* lncg = (const float*)d_in[28], *lncb = (const float*)d_in[29];
    const float* lnsg = (const float*)d_in[30], *lnsb = (const float*)d_in[31];
    const float* lnog = (const float*)d_in[32], *lnob = (const float*)d_in[33];
    const float* Wout = (const float*)d_in[34], *bout = (const float*)d_in[35];

    uintptr_t base = (uintptr_t)d_ws;
    auto alloc = [&](size_t bytes) -> void* {
        void* p = (void*)base;
        base += (bytes + 255) & ~(size_t)255;
        return p;
    };
    float*  xseq   = (float*)alloc((size_t)NSEQT * NBATCH * DMODEL * 4);
    float*  h      = (float*)alloc(NBATCH * DSTATE * 4);
    float*  Aflat  = (float*)alloc((size_t)NBATCH * DSTATE * DSTATE * 4);
    float*  Bflat  = (float*)alloc((size_t)NBATCH * DSTATE * DMODEL * 4);
    float*  rate   = (float*)alloc(NBATCH * 4);
    float*  fg     = (float*)alloc(NBATCH * DSTATE * 4);
    float*  ug     = (float*)alloc(NBATCH * DSTATE * 4);
    bf16_t* hidAbf = (bf16_t*)alloc(16 * DHID * 2);
    bf16_t* hidBbf = (bf16_t*)alloc(16 * DHID * 2);
    bf16_t* ysbf   = (bf16_t*)alloc((size_t)NBATCH * NSEQ * DMODEL * 2);
    bf16_t* pW2A   = (bf16_t*)alloc((size_t)DHID * DSTATE * DSTATE * 2);
    bf16_t* pW2B   = (bf16_t*)alloc((size_t)DHID * DSTATE * DMODEL * 2);
    bf16_t* pWout  = (bf16_t*)alloc((size_t)DMODEL * NVOCAB * 2);

    float* out_logits = (float*)d_out;
    float* out_h      = out_logits + (size_t)NBATCH * NSEQ * NVOCAB;

    // ---- one-time preparation ----
    {
        const long nInit = (long)NBATCH * DSTATE * DMODEL;
        init_state_kernel<<<(int)((nInit + 255) / 256), 256, 0, stream>>>(h, Aflat, Bflat, hidAbf, hidBbf);
        const long nEmb = (long)NSEQT * NBATCH * DMODEL;
        embed_kernel<<<(int)((nEmb + 255) / 256), 256, 0, stream>>>(ids, emb, persist, xseq);
        const long nA = (long)DHID * DSTATE * DSTATE;
        pack_bfrag_kernel<<<(int)((nA + 255) / 256), 256, 0, stream>>>(aW2, pW2A, DSTATE * DSTATE, DHID);
        const long nB = (long)DHID * DSTATE * DMODEL;
        pack_bfrag_kernel<<<(int)((nB + 255) / 256), 256, 0, stream>>>(bW2, pW2B, DSTATE * DMODEL, DHID);
        const long nW = (long)DMODEL * NVOCAB;
        pack_bfrag_kernel<<<(int)((nW + 255) / 256), 256, 0, stream>>>(Wout, pWout, NVOCAB, DMODEL);
    }

    // ---- the recurrence ----
    for (int t = 0; t < NSEQT; ++t) {
        ctrl_mlp_kernel<<<NBATCH, 256, 0, stream>>>(
            xseq, h, lncg, lncb,
            rW1, rb1, rW2, rb2, aW1, ab1, bW1, bb1,
            tW1, tb1, tW2, tb2, fW1, fb1, fW2, fb2, uW1, ub1, uW2, ub2,
            hidAbf, hidBbf, rate, fg, ug, t);
        hyp_wmma_kernel<DSTATE * DSTATE><<<(DSTATE * DSTATE / 16) / 8, 256, 0, stream>>>(
            hidAbf, pW2A, ab2, Aflat, rate);
        hyp_wmma_kernel<DSTATE * DMODEL><<<(DSTATE * DMODEL / 16) / 8, 256, 0, stream>>>(
            hidBbf, pW2B, bb2, Bflat, rate);
        state_out_kernel<<<NBATCH, 256, 0, stream>>>(
            Aflat, Bflat, xseq, h, fg, ug, lnsg, lnsb, Cmat, lnog, lnob, ysbf, t);
    }

    // ---- logits GEMM + h_final ----
    {
        logits_wmma_kernel<<<128 * 63, 256, 0, stream>>>(ysbf, pWout, bout, out_logits);
        hfinal_kernel<<<4, 256, 0, stream>>>(h, out_h);
    }
}
